// ConformerRNNTModel_84404697301337
// MI455X (gfx1250) — matrix operand order repaired and checked by reading, not compiled, not run on hardware
//
#include <hip/hip_runtime.h>
#include <hip/hip_bf16.h>
#include <math.h>

// ---------------- Conformer-RNNT forward for gfx1250 (MI455X) ----------------
// GEMMs use v_wmma_f32_16x16x32_f16. Wave tile = 16(M)x32(N): one A fragment
// feeds two WMMAs per K-step. K is a template parameter ({128,512}) so the
// K loop fully unrolls and the compiler pipelines global_load_b128 +
// v_cvt_pk_f16_f32 against the XDL (TRANS-class) WMMA pipe.
// A/B fragments use contiguous K (k = 16*hi + j); C layout per CDNA5 ISA 7.12.2.

typedef __attribute__((ext_vector_type(16))) _Float16 v16h;
typedef __attribute__((ext_vector_type(8)))  float    v8f;

constexpr int cB  = 4;
constexpr int cT  = 512;     // encoder time after 2x stride-2
constexpr int cD  = 128;
constexpr int cH  = 8;
constexpr int cV  = 100;
constexpr int cU  = 128;
constexpr int cFF = 512;
constexpr float cEPS = 1e-5f;

// ---------------------------------------------------------------- WMMA GEMM
// C[M,N] = act(A[M,K] @ W[N,K]^T + bias) (+ res + rs*val).  M % 16 == 0.
template<int ACT, int KK>
__global__ __launch_bounds__(128)
void k_gemm(const float* __restrict__ A, int lda,
            const float* __restrict__ W, int ldw,
            const float* __restrict__ bias,
            const float* __restrict__ res, float rs,
            float* __restrict__ C, int ldc,
            int M, int N)
{
    const int wave = threadIdx.x >> 5;
    const int lane = threadIdx.x & 31;
    const int tn = (blockIdx.x * 4 + wave) * 32;   // N tile base (32 wide)
    const int tm = blockIdx.y * 16;                // M tile base
    if (tn >= N) return;                           // uniform per wave

    const int r  = lane & 15;
    const int hi = lane >> 4;
    int w0 = tn + r;      if (w0 > N - 1) w0 = N - 1;   // clamp ragged N tiles
    int w1 = tn + 16 + r; if (w1 > N - 1) w1 = N - 1;

    const float* arow = A + (size_t)(tm + r) * lda + hi * 16;  // A: K = hi*16+j (contig 16)
    const float* wr0  = W + (size_t)w0 * ldw + hi * 16;        // B: K = hi*16+j (contig 16)
    const float* wr1  = W + (size_t)w1 * ldw + hi * 16;

    if (res) __builtin_prefetch(res + (size_t)(tm + r) * ldc + tn, 0, 1);

    v8f acc0 = {}, acc1 = {};
#pragma unroll
    for (int k0 = 0; k0 < KK; k0 += 32) {
        v16h af, b0, b1;
#pragma unroll
        for (int j = 0; j < 16; ++j) {
            af[j]     = (_Float16)arow[k0 + j];
        }
#pragma unroll
        for (int j = 0; j < 16; ++j) {
            b0[j] = (_Float16)wr0[k0 + j];
            b1[j] = (_Float16)wr1[k0 + j];
        }
        acc0 = __builtin_amdgcn_wmma_f32_16x16x32_f16(false, af, false, b0,
                                                      (short)0, acc0, false, false);
        acc1 = __builtin_amdgcn_wmma_f32_16x16x32_f16(false, af, false, b1,
                                                      (short)0, acc1, false, false);
    }

    // epilogue: C layout — VGPR i, lane l -> row = tm + i + 8*hi, col = base + (l&15)
    auto emit = [&](const v8f& acc, int colbase) {
        const int col = colbase + r;
        if (col >= N) return;
        const float bv = bias ? bias[col] : 0.f;
        if (res) {
#pragma unroll
            for (int i = 0; i < 8; ++i) {
                const int row = tm + hi * 8 + i;
                float v = acc[i] + bv;
                if (ACT == 1) v = fmaxf(v, 0.f);
                C[(size_t)row * ldc + col] = res[(size_t)row * ldc + col] + rs * v;
            }
        } else {
#pragma unroll
            for (int i = 0; i < 8; ++i) {
                const int row = tm + hi * 8 + i;
                float v = acc[i] + bv;
                if (ACT == 1) v = fmaxf(v, 0.f);
                C[(size_t)row * ldc + col] = v;
            }
        }
    };
    emit(acc0, tn);
    emit(acc1, tn + 16);
}

// ------------------------------------------------------------ subsampling convs
__global__ void k_subconv1(const float* __restrict__ src, const float* __restrict__ w,
                           const float* __restrict__ b, float* __restrict__ out)
{
    int idx = blockIdx.x * 256 + threadIdx.x;        // [B=4][O=128][T=1024]
    if (idx >= 4 * 128 * 1024) return;
    int t = idx & 1023, o = (idx >> 10) & 127, bb = idx >> 17;
    float acc = b[o];
#pragma unroll
    for (int i = 0; i < 2; ++i)
#pragma unroll
        for (int j = 0; j < 3; ++j) {
            int ti = 2 * t + j - 1;
            if (ti >= 0 && ti < 2048)
                acc += src[(bb * 2 + i) * 2048 + ti] * w[(o * 2 + i) * 3 + j];
        }
    out[idx] = acc > 0.f ? acc : 0.f;
}

__global__ void k_subconv2(const float* __restrict__ x, const float* __restrict__ w,
                           const float* __restrict__ b, float* __restrict__ out)
{
    int idx = blockIdx.x * 256 + threadIdx.x;        // [B=4][O=128][T=512]
    if (idx >= 4 * 128 * 512) return;
    int t = idx & 511, o = (idx >> 9) & 127, bb = idx >> 16;
    float acc = b[o];
#pragma unroll
    for (int j = 0; j < 3; ++j) {
        int ti = 2 * t + j - 1;
        if (ti < 0 || ti >= 1024) continue;
        const float* xp = x + (size_t)(bb * 128) * 1024 + ti;
        const float* wp = w + o * 128 * 3 + j;
        for (int i = 0; i < 128; ++i)
            acc += xp[i * 1024] * wp[i * 3];
    }
    out[idx] = acc > 0.f ? acc : 0.f;
}

// --------------------------------------------------------- transpose + layernorm
__global__ __launch_bounds__(128)
void k_trans_ln(const float* __restrict__ x2, const float* __restrict__ g,
                const float* __restrict__ be, float* __restrict__ out)
{
    int row = blockIdx.x;                // b*512 + t
    int b = row >> 9, t = row & 511;
    int d = threadIdx.x;
    float v = x2[((size_t)(b * 128 + d)) * 512 + t];
    __shared__ float red[128];
    red[d] = v; __syncthreads();
    for (int s = 64; s > 0; s >>= 1) { if (d < s) red[d] += red[d + s]; __syncthreads(); }
    float mean = red[0] * (1.f / 128.f); __syncthreads();
    float c = v - mean;
    red[d] = c * c; __syncthreads();
    for (int s = 64; s > 0; s >>= 1) { if (d < s) red[d] += red[d + s]; __syncthreads(); }
    float var = red[0] * (1.f / 128.f);
    out[(size_t)row * 128 + d] = c * rsqrtf(var + cEPS) * g[d] + be[d];
}

// LN(x [+ res]) over last dim 128, one row per block
__global__ __launch_bounds__(128)
void k_ln(const float* __restrict__ x, const float* __restrict__ res,
          const float* __restrict__ g, const float* __restrict__ be,
          float* __restrict__ out)
{
    int row = blockIdx.x, d = threadIdx.x;
    float v = x[(size_t)row * 128 + d];
    if (res) v += res[(size_t)row * 128 + d];
    __shared__ float red[128];
    red[d] = v; __syncthreads();
    for (int s = 64; s > 0; s >>= 1) { if (d < s) red[d] += red[d + s]; __syncthreads(); }
    float mean = red[0] * (1.f / 128.f); __syncthreads();
    float c = v - mean;
    red[d] = c * c; __syncthreads();
    for (int s = 64; s > 0; s >>= 1) { if (d < s) red[d] += red[d + s]; __syncthreads(); }
    float var = red[0] * (1.f / 128.f);
    out[(size_t)row * 128 + d] = c * rsqrtf(var + cEPS) * g[d] + be[d];
}

// ------------------------------------------------------------------- attention
// one block per (q, h, b); qkv row layout [.. 384]: Q[0:128] K[128:256] V[256:384]
__global__ __launch_bounds__(128)
void k_attn(const float* __restrict__ qkv, float* __restrict__ out)
{
    const int q = blockIdx.x, h = blockIdx.y, b = blockIdx.z;
    const int tid = threadIdx.x;
    __shared__ float s[512];
    __shared__ float red[128];
    __shared__ float part[128];

    float qv[16];
    const float* qp = qkv + ((size_t)(b * 512 + q)) * 384 + h * 16;
#pragma unroll
    for (int d = 0; d < 16; ++d) qv[d] = qp[d];

    for (int kk = tid; kk < 512; kk += 128) {
        const float* kp = qkv + ((size_t)(b * 512 + kk)) * 384 + 128 + h * 16;
        float acc = 0.f;
#pragma unroll
        for (int d = 0; d < 16; ++d) acc += qv[d] * kp[d];
        s[kk] = acc * 0.25f;                 // 1/sqrt(16)
    }
    float m = -1e30f;
    for (int kk = tid; kk < 512; kk += 128) m = fmaxf(m, s[kk]);
    red[tid] = m; __syncthreads();
    for (int st = 64; st > 0; st >>= 1) { if (tid < st) red[tid] = fmaxf(red[tid], red[tid + st]); __syncthreads(); }
    float mx = red[0]; __syncthreads();
    float lsum = 0.f;
    for (int kk = tid; kk < 512; kk += 128) { float e = __expf(s[kk] - mx); s[kk] = e; lsum += e; }
    red[tid] = lsum; __syncthreads();
    for (int st = 64; st > 0; st >>= 1) { if (tid < st) red[tid] += red[tid + st]; __syncthreads(); }
    float inv = 1.f / red[0];

    const int d = tid & 15, ch = tid >> 4;
    float acc = 0.f;
    for (int kk = ch * 64; kk < ch * 64 + 64; ++kk)
        acc += s[kk] * qkv[((size_t)(b * 512 + kk)) * 384 + 256 + h * 16 + d];
    part[tid] = acc; __syncthreads();
    if (tid < 16) {
        float o = 0.f;
#pragma unroll
        for (int cc = 0; cc < 8; ++cc) o += part[cc * 16 + tid];
        out[((size_t)(b * 512 + q)) * 128 + h * 16 + tid] = o * inv;
    }
}

// ----------------------------------------------------------------- conv module
__global__ void k_glu(const float* __restrict__ u, float* __restrict__ out)
{
    int idx = blockIdx.x * 256 + threadIdx.x;        // 2048*128
    if (idx >= 2048 * 128) return;
    int c = idx & 127, row = idx >> 7;
    float a = u[(size_t)row * 256 + c];
    float g = u[(size_t)row * 256 + 128 + c];
    out[idx] = a * (1.f / (1.f + __expf(-g)));
}

__global__ void k_dwconv_bn_silu(const float* __restrict__ in, const float* __restrict__ w,
                                 const float* __restrict__ bias,
                                 const float* __restrict__ bn_g, const float* __restrict__ bn_b,
                                 const float* __restrict__ bn_m, const float* __restrict__ bn_v,
                                 float* __restrict__ out)
{
    int idx = blockIdx.x * 256 + threadIdx.x;        // [B][T=512][C=128]
    if (idx >= 4 * 512 * 128) return;
    int c = idx & 127, t = (idx >> 7) & 511, b = idx >> 16;
    float acc = bias[c];
#pragma unroll
    for (int j = 0; j < 31; ++j) {
        int ti = t + j - 15;
        if (ti >= 0 && ti < 512)
            acc += in[((size_t)(b * 512 + ti)) * 128 + c] * w[c * 31 + j];
    }
    float y = (acc - bn_m[c]) * rsqrtf(bn_v[c] + cEPS) * bn_g[c] + bn_b[c];
    out[idx] = y * (1.f / (1.f + __expf(-y)));        // SiLU
}

// ------------------------------------------------------------------- predictor
__global__ void k_embed(const int* __restrict__ tgt, const float* __restrict__ emb,
                        float* __restrict__ out)
{
    int row = blockIdx.x, d = threadIdx.x;            // 512 rows (b*128+u)
    out[(size_t)row * 128 + d] = emb[(size_t)tgt[row] * 128 + d];
}

__global__ void k_bsum(const float* __restrict__ a, const float* __restrict__ b,
                       float* __restrict__ o)
{
    int i = blockIdx.x * 256 + threadIdx.x;
    if (i < 512) o[i] = a[i] + b[i];
}

__global__ __launch_bounds__(512)
void k_lstm(const float* __restrict__ xg, const float* __restrict__ whh,
            float* __restrict__ hs)
{
    __shared__ float h[4][128], c[4][128], gt[4][512];
    const int t = threadIdx.x;
    if (t < 128)
        for (int b = 0; b < 4; ++b) { h[b][t] = 0.f; c[b][t] = 0.f; }
    __syncthreads();
    const float* wr = whh + (size_t)t * 128;
    for (int u = 0; u < 128; ++u) {
        float a0 = xg[((size_t)(0 * 128 + u)) * 512 + t];
        float a1 = xg[((size_t)(1 * 128 + u)) * 512 + t];
        float a2 = xg[((size_t)(2 * 128 + u)) * 512 + t];
        float a3 = xg[((size_t)(3 * 128 + u)) * 512 + t];
        for (int j = 0; j < 128; ++j) {
            float w = wr[j];
            a0 += w * h[0][j]; a1 += w * h[1][j];
            a2 += w * h[2][j]; a3 += w * h[3][j];
        }
        gt[0][t] = a0; gt[1][t] = a1; gt[2][t] = a2; gt[3][t] = a3;
        __syncthreads();
        const int b = t >> 7, j = t & 127;
        float ig = gt[b][j], fg = gt[b][j + 128], gg = gt[b][j + 256], og = gt[b][j + 384];
        float si = 1.f / (1.f + __expf(-ig));
        float sf = 1.f / (1.f + __expf(-fg));
        float so = 1.f / (1.f + __expf(-og));
        float cc = sf * c[b][j] + si * tanhf(gg);
        float hh = so * tanhf(cc);
        c[b][j] = cc; h[b][j] = hh;
        hs[((size_t)(b * 128 + u)) * 128 + j] = hh;
        __syncthreads();
    }
}

// --------------------------------------------------------------------- joint
__global__ __launch_bounds__(128)
void k_joint(const float* __restrict__ e, const float* __restrict__ p,
             const float* __restrict__ fcb, float* __restrict__ out)
{
    const int n = blockIdx.x;             // (b*T + t)*U + u, 262144 blocks
    const int v = threadIdx.x;
    if (v >= cV) return;
    const int u  = n & 127;
    const int bt = n >> 7;
    const int b  = bt >> 9;
    out[(size_t)n * cV + v] = e[(size_t)bt * cV + v]
                            + p[((size_t)(b * 128 + u)) * cV + v]
                            + fcb[v];
}

// =============================================================== host launcher
extern "C" void kernel_launch(void* const* d_in, const int* in_sizes, int n_in,
                              void* d_out, int out_size, void* d_ws, size_t ws_size,
                              hipStream_t stream)
{
    (void)in_sizes; (void)n_in; (void)out_size; (void)ws_size;

    // ---- inputs: setup_inputs() flattened in dict insertion order ----
    const float* src      = (const float*)d_in[0];   // [4,2,2048]
    const int*   tgt      = (const int*)  d_in[1];   // [4,128]
    const float* sub_w1   = (const float*)d_in[2];
    const float* sub_b1   = (const float*)d_in[3];
    const float* sub_w2   = (const float*)d_in[4];
    const float* sub_b2   = (const float*)d_in[5];
    const float* enc_lng  = (const float*)d_in[6];
    const float* enc_lnb  = (const float*)d_in[7];
    const float* ff1_w1   = (const float*)d_in[8];   // [6,512,128]
    const float* ff1_b1   = (const float*)d_in[9];
    const float* ff1_w2   = (const float*)d_in[10];  // [6,128,512]
    const float* ff1_b2   = (const float*)d_in[11];
    const float* ff2_w1   = (const float*)d_in[12];
    const float* ff2_b1   = (const float*)d_in[13];
    const float* ff2_w2   = (const float*)d_in[14];
    const float* ff2_b2   = (const float*)d_in[15];
    const float* at_inw   = (const float*)d_in[16];  // [6,384,128]
    const float* at_inb   = (const float*)d_in[17];
    const float* at_ow    = (const float*)d_in[18];  // [6,128,128]
    const float* at_ob    = (const float*)d_in[19];
    const float* at_lng   = (const float*)d_in[20];
    const float* at_lnb   = (const float*)d_in[21];
    const float* cv_lng   = (const float*)d_in[22];
    const float* cv_lnb   = (const float*)d_in[23];
    const float* pw1_w    = (const float*)d_in[24];  // [6,256,128]
    const float* pw1_b    = (const float*)d_in[25];
    const float* dw_w     = (const float*)d_in[26];  // [6,128,31]
    const float* dw_b     = (const float*)d_in[27];
    const float* bn_g     = (const float*)d_in[28];
    const float* bn_b     = (const float*)d_in[29];
    const float* bn_m     = (const float*)d_in[30];
    const float* bn_v     = (const float*)d_in[31];
    const float* pw2_w    = (const float*)d_in[32];  // [6,128,128]
    const float* pw2_b    = (const float*)d_in[33];
    const float* blk_lng  = (const float*)d_in[34];
    const float* blk_lnb  = (const float*)d_in[35];
    const float* embed    = (const float*)d_in[36];  // [100,128]
    const float* w_ih     = (const float*)d_in[37];  // [512,128]
    const float* w_hh     = (const float*)d_in[38];  // [512,128]
    const float* b_ih     = (const float*)d_in[39];
    const float* b_hh     = (const float*)d_in[40];
    const float* fc_w     = (const float*)d_in[41];  // [100,256]
    const float* fc_b     = (const float*)d_in[42];

    // ---- workspace layout (floats), ~16.2 MB total ----
    float* wsf = (float*)d_ws;
    size_t off = 0;
    auto alloc = [&](size_t n) { float* p = wsf + off; off += n; return p; };
    float* X1  = alloc(4 * 128 * 1024);
    float* X2  = alloc(4 * 128 * 512);
    float* XA  = alloc(2048 * 128);      // encoder state [B*T, D]
    float* T0  = alloc(2048 * 512);      // FF hidden / pw1 out
    float* TQ  = alloc(2048 * 384);      // qkv
    float* T2  = alloc(2048 * 128);
    float* T3  = alloc(2048 * 128);
    float* EMB = alloc(512 * 128);
    float* XG  = alloc(512 * 512);
    float* HS  = alloc(512 * 128);
    float* EB  = alloc(2048 * 100);
    float* PB  = alloc(512 * 100);
    float* BS  = alloc(512);

    const int BT = cB * cT;              // 2048

    auto gemm = [&](const float* A, int lda, const float* W, int ldw,
                    const float* bias, const float* res, float rs,
                    float* C, int ldc, int M, int N, int K, int act) {
        dim3 g((N + 127) / 128, M / 16);   // 4 waves/block, 32 cols/wave
        dim3 b(128);
        if (K == 512) {
            k_gemm<0, 512><<<g, b, 0, stream>>>(A, lda, W, ldw, bias, res, rs, C, ldc, M, N);
        } else if (act) {
            k_gemm<1, 128><<<g, b, 0, stream>>>(A, lda, W, ldw, bias, res, rs, C, ldc, M, N);
        } else {
            k_gemm<0, 128><<<g, b, 0, stream>>>(A, lda, W, ldw, bias, res, rs, C, ldc, M, N);
        }
    };

    // ---- subsampling ----
    k_subconv1<<<(4 * 128 * 1024 + 255) / 256, 256, 0, stream>>>(src, sub_w1, sub_b1, X1);
    k_subconv2<<<(4 * 128 * 512 + 255) / 256, 256, 0, stream>>>(X1, sub_w2, sub_b2, X2);
    k_trans_ln<<<BT, 128, 0, stream>>>(X2, enc_lng, enc_lnb, XA);

    // ---- conformer blocks ----
    for (int l = 0; l < 6; ++l) {
        // FF1: x += 0.5 * ff(x)
        gemm(XA, 128, ff1_w1 + (size_t)l * cFF * cD, 128, ff1_b1 + l * cFF,
             nullptr, 0.f, T0, 512, BT, 512, 128, 1);
        gemm(T0, 512, ff1_w2 + (size_t)l * cD * cFF, 512, ff1_b2 + l * cD,
             XA, 0.5f, XA, 128, BT, 128, 512, 0);
        // MHA
        gemm(XA, 128, at_inw + (size_t)l * 384 * 128, 128, at_inb + l * 384,
             nullptr, 0.f, TQ, 384, BT, 384, 128, 0);
        k_attn<<<dim3(cT, cH, cB), 128, 0, stream>>>(TQ, T2);
        gemm(T2, 128, at_ow + (size_t)l * 128 * 128, 128, at_ob + l * 128,
             nullptr, 0.f, T3, 128, BT, 128, 128, 0);
        k_ln<<<BT, 128, 0, stream>>>(XA, T3, at_lng + l * 128, at_lnb + l * 128, XA);
        // conv module: x += conv(x)
        k_ln<<<BT, 128, 0, stream>>>(XA, nullptr, cv_lng + l * 128, cv_lnb + l * 128, T3);
        gemm(T3, 128, pw1_w + (size_t)l * 256 * 128, 128, pw1_b + l * 256,
             nullptr, 0.f, T0, 256, BT, 256, 128, 0);
        k_glu<<<(BT * 128 + 255) / 256, 256, 0, stream>>>(T0, T2);
        k_dwconv_bn_silu<<<(BT * 128 + 255) / 256, 256, 0, stream>>>(
            T2, dw_w + (size_t)l * 128 * 31, dw_b + l * 128,
            bn_g + l * 128, bn_b + l * 128, bn_m + l * 128, bn_v + l * 128, T3);
        gemm(T3, 128, pw2_w + (size_t)l * 128 * 128, 128, pw2_b + l * 128,
             XA, 1.0f, XA, 128, BT, 128, 128, 0);
        // FF2: x += 0.5 * ff(x)
        gemm(XA, 128, ff2_w1 + (size_t)l * cFF * cD, 128, ff2_b1 + l * cFF,
             nullptr, 0.f, T0, 512, BT, 512, 128, 1);
        gemm(T0, 512, ff2_w2 + (size_t)l * cD * cFF, 512, ff2_b2 + l * cD,
             XA, 0.5f, XA, 128, BT, 128, 512, 0);
        k_ln<<<BT, 128, 0, stream>>>(XA, nullptr, blk_lng + l * 128, blk_lnb + l * 128, XA);
    }

    // ---- predictor (LSTM) ----
    k_embed<<<cB * cU, 128, 0, stream>>>(tgt, embed, EMB);
    k_bsum<<<2, 256, 0, stream>>>(b_ih, b_hh, BS);
    gemm(EMB, 128, w_ih, 128, BS, nullptr, 0.f, XG, 512, cB * cU, 512, 128, 0);
    k_lstm<<<1, 512, 0, stream>>>(XG, w_hh, HS);

    // ---- joint ----
    gemm(XA, 128, fc_w,       256, nullptr, nullptr, 0.f, EB, 100, BT,      100, 128, 0);
    gemm(HS, 128, fc_w + 128, 256, nullptr, nullptr, 0.f, PB, 100, cB * cU, 100, 128, 0);
    k_joint<<<BT * cU, 128, 0, stream>>>(EB, PB, fc_b, (float*)d_out);
}
